// NLayerGAT_12601434046864
// MI455X (gfx1250) — compile-verified, matrix operand order checked by reading, not run on hardware
//
#include <hip/hip_runtime.h>
#include <math.h>

// ---------------------------------------------------------------------------
// 3-layer GAT for MI455X (gfx1250).
//   GEMMs:  v_wmma_f32_16x16x32_bf16 (bf16 inputs, f32 accumulate)
//   Edge softmax + aggregation: L2-resident atomics (entire h table ~12.8MB
//   fits in the 192MB L2, so random src/dst gathers and scatter atomic adds
//   stay on-chip).
// ---------------------------------------------------------------------------

typedef __bf16 bf16_t;
typedef __attribute__((ext_vector_type(16))) __bf16 v16bf;
typedef __attribute__((ext_vector_type(8)))  float  v8f;

#define NEG_SLOPE 0.2f

// --- ISA 7.12.2: 16-bit A/B fragment K index mapping -----------------------
// element i of v16bf in lane l:  reg r = i>>1, half p = i&1
//   lanes 0-15 : K = (r<4 ? 2r : 16+2(r-4)) + p
//   lanes 16-31: same + 8
__device__ __forceinline__ int kmap(int r, int p, int hi) {
  int base = (r < 4) ? (2 * r) : (16 + 2 * (r - 4));
  return base + (hi ? 8 : 0) + p;
}

// ---------------------------------------------------------------------------
// GEMM: H[n, DOUT_] = X[n, DIN] @ W[DIN, DOUT_]   (one wave per 16x16 tile)
// ---------------------------------------------------------------------------
template <int DIN, int DOUT_>
__global__ void gat_gemm_wmma(const float* __restrict__ X,
                              const float* __restrict__ W,
                              float* __restrict__ H,
                              int totalWaves) {
  int wave = (int)((blockIdx.x * blockDim.x + threadIdx.x) >> 5);
  if (wave >= totalWaves) return;          // wave-uniform: EXEC stays all-1s
  int lane = threadIdx.x & 31;
  constexpr int NTN = DOUT_ / 16;
  int mtile = wave / NTN;
  int ntile = wave % NTN;
  int row = mtile * 16 + (lane & 15);      // A: lane -> M row
  int col = ntile * 16 + (lane & 15);      // B: lane -> N col
  int hi  = lane >> 4;

  v8f acc = {};
#pragma unroll
  for (int k0 = 0; k0 < DIN; k0 += 32) {
    v16bf a, b;
#pragma unroll
    for (int i = 0; i < 16; ++i) {
      int k = k0 + kmap(i >> 1, i & 1, hi);
      a[i] = (bf16_t)X[row * DIN + k];
      b[i] = (bf16_t)W[k * DOUT_ + col];
    }
    acc = __builtin_amdgcn_wmma_f32_16x16x32_bf16(
        /*neg_a=*/false, a, /*neg_b=*/false, b,
        /*c_mod=*/(short)0, acc, /*reuse_a=*/false, /*reuse_b=*/false);
  }

  // C/D layout: VGPR v, lanes 0-15 -> M=v, lanes 16-31 -> M=v+8; N = lane&15
#pragma unroll
  for (int v = 0; v < 8; ++v) {
    H[(mtile * 16 + hi * 8 + v) * DOUT_ + col] = acc[v];
  }
}

// ---------------------------------------------------------------------------
// Per-node attention logits: asrc[i] = h[i].a_s ; adst[i] = h[i].a_d
// ---------------------------------------------------------------------------
template <int DOUT_>
__global__ void alpha_kernel(const float* __restrict__ H,
                             const float* __restrict__ a_s,
                             const float* __restrict__ a_d,
                             float* __restrict__ asrc,
                             float* __restrict__ adst, int n) {
  int i = blockIdx.x * blockDim.x + threadIdx.x;
  if (i >= n) return;
  float s = 0.f, d = 0.f;
#pragma unroll
  for (int j = 0; j < DOUT_; ++j) {
    float h = H[i * DOUT_ + j];
    s += h * a_s[j];
    d += h * a_d[j];
  }
  asrc[i] = s;
  adst[i] = d;
}

// ---------------------------------------------------------------------------
// Monotone float <-> uint key for atomic max
// ---------------------------------------------------------------------------
__device__ __forceinline__ unsigned fkey(float f) {
  unsigned u = __float_as_uint(f);
  return (u & 0x80000000u) ? ~u : (u | 0x80000000u);
}
__device__ __forceinline__ float keyf(unsigned k) {
  unsigned u = (k & 0x80000000u) ? (k & 0x7FFFFFFFu) : ~k;
  return __uint_as_float(u);
}

// zero accumulators / max keys / sums for one layer
__global__ void init_layer(float* __restrict__ outacc, unsigned* __restrict__ mkey,
                           float* __restrict__ ssum, int n, int dout) {
  int idx = blockIdx.x * blockDim.x + threadIdx.x;
  if (idx < n * dout) outacc[idx] = 0.f;
  if (idx < n) { mkey[idx] = 0u; ssum[idx] = 0.f; }
}

// Pass 1: e = leaky_relu(asrc[src] + adst[dst]); segment max into mkey[dst]
__global__ void edge_score_max(const int* __restrict__ ei,
                               const float* __restrict__ asrc,
                               const float* __restrict__ adst,
                               float* __restrict__ ebuf,
                               unsigned* __restrict__ mkey, int E, int etot) {
  int e = blockIdx.x * blockDim.x + threadIdx.x;
  if (e >= etot) return;
  int s, d;
  if (e < E) { s = ei[e]; d = ei[E + e]; } else { s = d = e - E; }
  float v = asrc[s] + adst[d];
  v = v > 0.f ? v : NEG_SLOPE * v;
  ebuf[e] = v;
  atomicMax(&mkey[d], fkey(v));
}

// Pass 2: ex = exp(e - m[dst]); segment sum into ssum[dst]
__global__ void edge_exp_sum(const int* __restrict__ ei,
                             float* __restrict__ ebuf,
                             const unsigned* __restrict__ mkey,
                             float* __restrict__ ssum, int E, int etot) {
  int e = blockIdx.x * blockDim.x + threadIdx.x;
  if (e >= etot) return;
  int d = (e < E) ? ei[E + e] : (e - E);
  float ex = __expf(ebuf[e] - keyf(mkey[d]));
  ebuf[e] = ex;
  atomicAdd(&ssum[d], ex);
}

// Pass 3: out[dst] += (ex/s[dst]) * h[src]   (one thread per edge-feature)
template <int DOUT_, int LG>
__global__ void edge_aggregate(const int* __restrict__ ei,
                               const float* __restrict__ H,
                               const float* __restrict__ ebuf,
                               const float* __restrict__ ssum,
                               float* __restrict__ outacc, int E, int etot) {
  long long idx = (long long)blockIdx.x * blockDim.x + threadIdx.x;
  if (idx >= ((long long)etot << LG)) return;
  int e = (int)(idx >> LG);
  int f = (int)(idx & (DOUT_ - 1));
  int s, d;
  if (e < E) { s = ei[e]; d = ei[E + e]; } else { s = d = e - E; }
  float w = ebuf[e] / ssum[d];
  atomicAdd(&outacc[(long long)d * DOUT_ + f], H[(long long)s * DOUT_ + f] * w);
}

// Pass 4: out = act(acc + bias)   act: 0 = relu, 1 = tanh
__global__ void finalize(const float* __restrict__ acc,
                         const float* __restrict__ bias,
                         float* __restrict__ out,
                         long long total, int dout, int act) {
  long long idx = (long long)blockIdx.x * blockDim.x + threadIdx.x;
  if (idx >= total) return;
  int f = (int)(idx % dout);
  float v = acc[idx] + bias[f];
  out[idx] = (act == 0) ? fmaxf(v, 0.f) : tanhf(v);
}

// ---------------------------------------------------------------------------
// Driver
// ---------------------------------------------------------------------------
extern "C" void kernel_launch(void* const* d_in, const int* in_sizes, int n_in,
                              void* d_out, int out_size, void* d_ws, size_t ws_size,
                              hipStream_t stream) {
  (void)n_in; (void)out_size; (void)ws_size;
  const float* x  = (const float*)d_in[0];
  const int*   ei = (const int*)d_in[1];
  const int n    = in_sizes[0] / 128;   // 50000
  const int E    = in_sizes[1] / 2;     // 800000
  const int etot = E + n;               // + self loops

  // workspace carve-out (floats): ~30 MB total
  const size_t N64 = (size_t)n * 64;
  float*    hbuf   = (float*)d_ws;          // N*64
  float*    outacc = hbuf + N64;            // N*64
  float*    asrc   = outacc + N64;          // N
  float*    adst   = asrc + n;              // N
  unsigned* mkey   = (unsigned*)(adst + n); // N
  float*    ssum   = (float*)(mkey + n);    // N
  float*    ebuf   = ssum + n;              // E + N

  const int T = 256;
  const float* cur = x;

  for (int l = 0; l < 3; ++l) {
    const float* Wl  = (const float*)d_in[2 + 4 * l];
    const float* asw = (const float*)d_in[3 + 4 * l];
    const float* adw = (const float*)d_in[4 + 4 * l];
    const float* bl  = (const float*)d_in[5 + 4 * l];
    const int dout = (l == 2) ? 32 : 64;

    // GEMM (WMMA) + logits
    const int waves  = (n / 16) * (dout / 16);
    const int gblks  = (waves * 32 + T - 1) / T;
    if (l == 0)      gat_gemm_wmma<128, 64><<<gblks, T, 0, stream>>>(cur, Wl, hbuf, waves);
    else if (l == 1) gat_gemm_wmma< 64, 64><<<gblks, T, 0, stream>>>(cur, Wl, hbuf, waves);
    else             gat_gemm_wmma< 64, 32><<<gblks, T, 0, stream>>>(cur, Wl, hbuf, waves);

    if (dout == 64) alpha_kernel<64><<<(n + T - 1) / T, T, 0, stream>>>(hbuf, asw, adw, asrc, adst, n);
    else            alpha_kernel<32><<<(n + T - 1) / T, T, 0, stream>>>(hbuf, asw, adw, asrc, adst, n);

    // segment softmax + aggregate
    init_layer<<<((size_t)n * dout + T - 1) / T, T, 0, stream>>>(outacc, mkey, ssum, n, dout);
    edge_score_max<<<(etot + T - 1) / T, T, 0, stream>>>(ei, asrc, adst, ebuf, mkey, E, etot);
    edge_exp_sum<<<(etot + T - 1) / T, T, 0, stream>>>(ei, ebuf, mkey, ssum, E, etot);

    const long long tot = (long long)etot * dout;
    const int ablks = (int)((tot + T - 1) / T);
    if (dout == 64) edge_aggregate<64, 6><<<ablks, T, 0, stream>>>(ei, hbuf, ebuf, ssum, outacc, E, etot);
    else            edge_aggregate<32, 5><<<ablks, T, 0, stream>>>(ei, hbuf, ebuf, ssum, outacc, E, etot);

    float* target = (l == 2) ? (float*)d_out : outacc;
    const long long ntot = (long long)n * dout;
    finalize<<<(int)((ntot + T - 1) / T), T, 0, stream>>>(outacc, bl, target, ntot, dout, (l == 2) ? 1 : 0);

    cur = outacc;  // activated output feeds next layer's GEMM (stream-ordered)
  }
}
